// WordToSentence_57698590655215
// MI455X (gfx1250) — compile-verified, matrix operand order checked by reading, not run on hardware
//
#include <hip/hip_runtime.h>
#include <cstdint>
#include <cstddef>

#define B_   2048
#define L_   64
#define E_   200
#define EP_  224     // E padded to multiple of 32 (WMMA K step)
#define H_   128
#define H3_  384
#define D2_  256     // 2H
#define C_   128
#define BL_  (B_ * L_)

typedef _Float16 half_t;
typedef __attribute__((ext_vector_type(8)))  _Float16     v8h;
typedef __attribute__((ext_vector_type(16))) _Float16     v16h;
typedef __attribute__((ext_vector_type(8)))  float        v8f;
typedef __attribute__((ext_vector_type(4)))  unsigned int u32x4;
typedef __attribute__((ext_vector_type(8)))  int          i32x8;
typedef __attribute__((ext_vector_type(4)))  int          i32x4;

// ---- WMMA fragment helpers (16x32 f16 A/B fragments, row-major source) ----
// A-fragment layout (ISA 7.12.2): lane = row%16; halves 0..7 = K{0..7|8..15},
// halves 8..15 = K{16..23|24..31} for lanes <16 | >=16.  B[k][n] = W[n][k]
// with W row-major means the SAME loader works with n as the lane-row.
__device__ __forceinline__ v16h frag_load(const half_t* p, int ld, int row0,
                                          int k0, int lane) {
  const half_t* q = p + (size_t)(row0 + (lane & 15)) * ld + (k0 + ((lane >> 4) << 3));
  v8h lo = *(const v8h*)q;          // K+0..7   (16B aligned: ld,k0 mult of 8)
  v8h hi = *(const v8h*)(q + 16);   // K+16..23
  return __builtin_shufflevector(lo, hi, 0,1,2,3,4,5,6,7,8,9,10,11,12,13,14,15);
}

// ---- generic WMMA GEMM:  C[M,N] = A[M,K] * W[N,K]^T + bias[N]  (f16 in, f32 out)
// one 16x16 output tile per wave, 8 waves/block, grid=(Mtiles/8, Ntiles)
__global__ __launch_bounds__(256) void wmma_gemm_bias(
    const half_t* __restrict__ A, const half_t* __restrict__ W,
    const float* __restrict__ bias, float* __restrict__ Cm,
    int ld, int K, int N) {
  int lane = threadIdx.x & 31;
  int wave = threadIdx.x >> 5;
  int row0 = (blockIdx.x * 8 + wave) * 16;
  int col0 = blockIdx.y * 16;
  v8f acc = {};
  for (int k0 = 0; k0 < K; k0 += 32) {
    v16h a = frag_load(A, ld, row0, k0, lane);
    v16h b = frag_load(W, ld, col0, k0, lane);
    acc = __builtin_amdgcn_wmma_f32_16x16x32_f16(false, a, false, b,
                                                 (short)0, acc, false, false);
  }
  float bv = bias[col0 + (lane & 15)];
  int n  = col0 + (lane & 15);
  int mb = row0 + ((lane >> 4) << 3);   // C layout: vgpr j -> row mb+j, col n
#pragma unroll
  for (int j = 0; j < 8; ++j) Cm[(size_t)(mb + j) * N + n] = acc[j] + bv;
}

// ---- weight packing / conversion ----
__global__ void pack_wih(const float* __restrict__ Wf, const float* __restrict__ Wb,
                         half_t* __restrict__ dst) {
  int idx = blockIdx.x * blockDim.x + threadIdx.x;
  if (idx >= 2 * H3_ * EP_) return;
  int d = idx / (H3_ * EP_);
  int rem = idx - d * (H3_ * EP_);
  int r = rem / EP_, k = rem - r * EP_;
  const float* W = d ? Wb : Wf;
  dst[idx] = (k < E_) ? (half_t)W[r * E_ + k] : (half_t)0.f;
}

__global__ void cvt_f16(const float* __restrict__ src, half_t* __restrict__ dst, int n) {
  int i = blockIdx.x * blockDim.x + threadIdx.x;
  if (i < n) dst[i] = (half_t)src[i];
}

__global__ void gather_emb(const int* __restrict__ x, const float* __restrict__ emb,
                           half_t* __restrict__ X16) {
  int idx = blockIdx.x * blockDim.x + threadIdx.x;
  if (idx >= BL_ * EP_) return;
  int r = idx / EP_, k = idx - r * EP_;
  int tok = x[r];
  X16[idx] = (k < E_) ? (half_t)emb[(size_t)tok * E_ + k] : (half_t)0.f;
}

// ---- persistent bidirectional GRU: 16 batch rows per block, 64 sequential steps
// per step: TDM async-load xp tile (16x384 f32, row stride L*384) to LDS while
// the 8 waves do hp = h @ Whh^T via 24 WMMA tiles; then fused gate math.
__global__ __launch_bounds__(256) void gru_kernel(
    const float* __restrict__ XPf, const float* __restrict__ XPb,
    const half_t* __restrict__ Whhf, const half_t* __restrict__ Whhb,
    const float* __restrict__ bhhf, const float* __restrict__ bhhb,
    const int* __restrict__ slen,
    float* __restrict__ OUT, half_t* __restrict__ OUT16) {
  const int dir = blockIdx.y;
  const int b0  = blockIdx.x * 16;
  const float*  XP  = dir ? XPb  : XPf;
  const half_t* Whh = dir ? Whhb : Whhf;
  const float*  bhh = dir ? bhhb : bhhf;

  __shared__ float  xp_s[16][H3_];   // 24 KB  (TDM destination)
  __shared__ float  hp_s[16][H3_];   // 24 KB
  __shared__ float  h_s [16][H_];    //  8 KB  (f32 recurrence state)
  __shared__ half_t h16_s[16][H_];   //  4 KB  (f16 copy for WMMA A)

  const int tid = threadIdx.x, lane = tid & 31, wave = tid >> 5;

  for (int i = tid; i < 16 * H_; i += 256) {
    h_s[i >> 7][i & 127] = 0.f;
    h16_s[i >> 7][i & 127] = (half_t)0.f;
  }
  __syncthreads();

  for (int s = 0; s < L_; ++s) {
    const int t = dir ? (L_ - 1 - s) : s;     // backward dir scans reversed time

    if (wave == 0) {
      // Tensor DMA descriptor (CDNA5 ISA): 2D tile 384 x 16, dtype 4B,
      // tensor_dim0_stride = L*384 elements, dest = xp_s.
      unsigned long long ga =
          (unsigned long long)(uintptr_t)(XP + ((size_t)b0 * L_ + t) * H3_);
      unsigned int la = (unsigned int)(uintptr_t)(&xp_s[0][0]);
      u32x4 g0;
      g0[0] = 1u;                                            // count=1
      g0[1] = la;                                            // lds_addr [63:32]
      g0[2] = (unsigned int)(ga & 0xFFFFFFFFull);            // global_addr lo
      g0[3] = (unsigned int)((ga >> 32) & 0x1FFFFFFull)      // global_addr[56:32]
            | (2u << 30);                                    // type=2 (image)
      i32x8 g1;
      g1[0] = 0x20000;             // data_size=2 (4 bytes) at [17:16]
      g1[1] = (int)(H3_ << 16);    // tensor_dim0 low16 at [63:48]
      g1[2] = (int)(16  << 16);    // tensor_dim0 hi=0 | tensor_dim1 low16
      g1[3] = (int)(H3_ << 16);    // tensor_dim1 hi=0 | tile_dim0=384
      g1[4] = 16;                  // tile_dim1=16, tile_dim2=0
      g1[5] = L_ * H3_;            // tensor_dim0_stride low32 = 24576
      g1[6] = 0;
      g1[7] = 0;
      i32x4 gz4 = {0, 0, 0, 0};
      i32x8 gz8 = {0, 0, 0, 0, 0, 0, 0, 0};
      // clang-23 / therock-10.0 form: 6 args (g0, g1, g2, g3, g4, cpol)
      __builtin_amdgcn_tensor_load_to_lds(g0, g1, gz4, gz4, gz8, 0);
    }

    // hp = h @ Whh^T + bhh : 24 N-tiles over 8 waves, K = 128 (4 WMMAs each)
#pragma unroll
    for (int tt = 0; tt < 3; ++tt) {
      int col0 = (wave + tt * 8) * 16;
      v8f acc = {};
#pragma unroll
      for (int k0 = 0; k0 < H_; k0 += 32) {
        v16h a = frag_load(&h16_s[0][0], H_, 0, k0, lane);   // ds_load_b128 x2
        v16h b = frag_load(Whh, H_, col0, k0, lane);
        acc = __builtin_amdgcn_wmma_f32_16x16x32_f16(false, a, false, b,
                                                     (short)0, acc, false, false);
      }
      int n  = col0 + (lane & 15);
      float bv = bhh[n];
      int mb = (lane >> 4) << 3;
#pragma unroll
      for (int j = 0; j < 8; ++j) hp_s[mb + j][n] = acc[j] + bv;
    }

    if (wave == 0) __builtin_amdgcn_s_wait_tensorcnt(0);
    __syncthreads();   // hp ready + xp tile landed in LDS

    // fused GRU gates (torch order r,z,n) + mask + dual-precision output
    for (int i = tid; i < 16 * H_; i += 256) {
      int r = i >> 7, c = i & 127;
      float xr = xp_s[r][c], xz = xp_s[r][c + H_], xn = xp_s[r][c + 2 * H_];
      float hr = hp_s[r][c], hz = hp_s[r][c + H_], hn = hp_s[r][c + 2 * H_];
      float rg = 1.f / (1.f + __expf(-(xr + hr)));
      float zg = 1.f / (1.f + __expf(-(xz + hz)));
      float ng = tanhf(xn + rg * hn);
      float h  = (1.f - zg) * ng + zg * h_s[r][c];
      h_s[r][c]   = h;
      h16_s[r][c] = (half_t)h;
      float m = (t < slen[b0 + r]) ? h : 0.f;           // sequence mask
      size_t o = ((size_t)(b0 + r) * L_ + t) * D2_ + dir * H_ + c;
      OUT[o]   = m;
      OUT16[o] = (half_t)m;
    }
    __syncthreads();   // h16 stable before next step's WMMA / TDM reissue
  }
}

// ---- BatchNorm1d(L) stats: mean/rstd over (B, C) per sequence position ----
__global__ __launch_bounds__(256) void bn_stats(const float* __restrict__ P,
                                                float* __restrict__ stats) {
  int l = blockIdx.x;
  float s1 = 0.f, s2 = 0.f;
  for (int i = threadIdx.x; i < B_ * C_; i += 256) {
    int b = i >> 7, c = i & 127;
    float v = P[((size_t)b * L_ + l) * C_ + c];
    s1 += v; s2 += v * v;
  }
#pragma unroll
  for (int off = 16; off > 0; off >>= 1) {
    s1 += __shfl_down(s1, off, 32);
    s2 += __shfl_down(s2, off, 32);
  }
  __shared__ float r1[8], r2[8];
  if ((threadIdx.x & 31) == 0) { r1[threadIdx.x >> 5] = s1; r2[threadIdx.x >> 5] = s2; }
  __syncthreads();
  if (threadIdx.x == 0) {
    float t1 = 0.f, t2 = 0.f;
    for (int w = 0; w < 8; ++w) { t1 += r1[w]; t2 += r2[w]; }
    const float inv = 1.f / (float)(B_ * C_);
    float mean = t1 * inv;
    float var  = t2 * inv - mean * mean;
    stats[l * 2]     = mean;
    stats[l * 2 + 1] = rsqrtf(var + 1e-5f);
  }
}

// ---- attention logits: relu(bn(proj)) . ctx ----
__global__ void logits_kernel(const float* __restrict__ P, const float* __restrict__ stats,
                              const float* __restrict__ gamma, const float* __restrict__ beta,
                              const float* __restrict__ ctx, float* __restrict__ att) {
  int idx = blockIdx.x * blockDim.x + threadIdx.x;
  if (idx >= BL_) return;
  int l = idx & (L_ - 1);
  float mean = stats[l * 2], rstd = stats[l * 2 + 1];
  float g = gamma[l], be = beta[l];
  const float* row = P + (size_t)idx * C_;
  float acc = 0.f;
  for (int c = 0; c < C_; ++c) {
    float v = (row[c] - mean) * rstd * g + be;
    acc += fmaxf(v, 0.f) * ctx[c];
  }
  att[idx] = acc;
}

// ---- softmax over L + weighted sum to [B, 2H] ----
__global__ __launch_bounds__(256) void finalize_kernel(const float* __restrict__ OUT,
                                                       const float* __restrict__ att,
                                                       float* __restrict__ out) {
  int b = blockIdx.x;
  __shared__ float lg[L_];
  if (threadIdx.x < L_) lg[threadIdx.x] = att[(size_t)b * L_ + threadIdx.x];
  __syncthreads();
  float mx = -3.4e38f;
  for (int l = 0; l < L_; ++l) mx = fmaxf(mx, lg[l]);
  float sum = 0.f;
  for (int l = 0; l < L_; ++l) sum += __expf(lg[l] - mx);
  float inv = 1.f / sum;
  int d = threadIdx.x;                       // 256 threads -> one output dim each
  float acc = 0.f;
  for (int l = 0; l < L_; ++l)
    acc += OUT[((size_t)b * L_ + l) * D2_ + d] * (__expf(lg[l] - mx) * inv);
  out[(size_t)b * D2_ + d] = acc;
}

extern "C" void kernel_launch(void* const* d_in, const int* in_sizes, int n_in,
                              void* d_out, int out_size, void* d_ws, size_t ws_size,
                              hipStream_t stream) {
  (void)in_sizes; (void)n_in; (void)out_size; (void)ws_size;
  const int*   x     = (const int*)d_in[0];
  const int*   slen  = (const int*)d_in[2];
  const float* emb   = (const float*)d_in[3];
  const float* Wih_f = (const float*)d_in[4];
  const float* Whh_f = (const float*)d_in[5];
  const float* bih_f = (const float*)d_in[6];
  const float* bhh_f = (const float*)d_in[7];
  const float* Wih_b = (const float*)d_in[8];
  const float* Whh_b = (const float*)d_in[9];
  const float* bih_b = (const float*)d_in[10];
  const float* bhh_b = (const float*)d_in[11];
  const float* Wp    = (const float*)d_in[12];
  const float* bp    = (const float*)d_in[13];
  const float* gamma = (const float*)d_in[14];
  const float* beta  = (const float*)d_in[15];
  const float* ctx   = (const float*)d_in[16];
  float* out = (float*)d_out;

  char* w = (char*)d_ws;
  auto alloc = [&](size_t bytes) {
    char* p = w; w += (bytes + 255) & ~(size_t)255; return p;
  };
  half_t* Wih16  = (half_t*)alloc((size_t)2 * H3_ * EP_ * sizeof(half_t));
  half_t* Whh16f = (half_t*)alloc((size_t)H3_ * H_ * sizeof(half_t));
  half_t* Whh16b = (half_t*)alloc((size_t)H3_ * H_ * sizeof(half_t));
  half_t* Wp16   = (half_t*)alloc((size_t)C_ * D2_ * sizeof(half_t));
  half_t* X16    = (half_t*)alloc((size_t)BL_ * EP_ * sizeof(half_t));
  float*  XPf    = (float*) alloc((size_t)BL_ * H3_ * sizeof(float));
  float*  XPb    = (float*) alloc((size_t)BL_ * H3_ * sizeof(float));
  float*  OUT    = (float*) alloc((size_t)BL_ * D2_ * sizeof(float));
  half_t* OUT16  = (half_t*)alloc((size_t)BL_ * D2_ * sizeof(half_t));
  float*  PROJ   = (float*) alloc((size_t)BL_ * C_ * sizeof(float));
  float*  stats  = (float*) alloc((size_t)L_ * 2 * sizeof(float));
  float*  att    = (float*) alloc((size_t)BL_ * sizeof(float));

  pack_wih<<<(2 * H3_ * EP_ + 255) / 256, 256, 0, stream>>>(Wih_f, Wih_b, Wih16);
  cvt_f16<<<(H3_ * H_ + 255) / 256, 256, 0, stream>>>(Whh_f, Whh16f, H3_ * H_);
  cvt_f16<<<(H3_ * H_ + 255) / 256, 256, 0, stream>>>(Whh_b, Whh16b, H3_ * H_);
  cvt_f16<<<(C_ * D2_ + 255) / 256, 256, 0, stream>>>(Wp, Wp16, C_ * D2_);
  gather_emb<<<(BL_ * EP_ + 255) / 256, 256, 0, stream>>>(x, emb, X16);

  dim3 blk(256);
  // xp = emb(x) @ Wih^T + bih (both directions; K padded to 224 with zeros)
  wmma_gemm_bias<<<dim3(BL_ / 16 / 8, H3_ / 16), blk, 0, stream>>>(
      X16, Wih16, bih_f, XPf, EP_, EP_, H3_);
  wmma_gemm_bias<<<dim3(BL_ / 16 / 8, H3_ / 16), blk, 0, stream>>>(
      X16, Wih16 + (size_t)H3_ * EP_, bih_b, XPb, EP_, EP_, H3_);
  // sequential recurrence (both directions via grid.y)
  gru_kernel<<<dim3(B_ / 16, 2), blk, 0, stream>>>(
      XPf, XPb, Whh16f, Whh16b, bhh_f, bhh_b, slen, OUT, OUT16);
  // proj = out @ Wp^T + bp
  wmma_gemm_bias<<<dim3(BL_ / 16 / 8, C_ / 16), blk, 0, stream>>>(
      OUT16, Wp16, bp, PROJ, D2_, D2_, C_);
  bn_stats<<<L_, blk, 0, stream>>>(PROJ, stats);
  logits_kernel<<<(BL_ + 255) / 256, 256, 0, stream>>>(PROJ, stats, gamma, beta, ctx, att);
  finalize_kernel<<<B_, blk, 0, stream>>>(OUT, att, out);
}
